// TransformerSeq2Seq_67078799229310
// MI455X (gfx1250) — compile-verified, matrix operand order checked by reading, not run on hardware
//
#include <hip/hip_runtime.h>
#include <cstdint>
#include <cstddef>

// ---------------- problem constants ----------------
#define D_MODEL 1024
#define NHEAD   16
#define HDIM    64
#define DFF_N   4096
#define BSZ     8
#define SEQ     512
#define MROWS   (BSZ*SEQ)   // 4096
#define MDIST   128

// ---------------- gfx1250 async global->LDS (guarded) ----------------
#if defined(__has_builtin)
#  if __has_builtin(__builtin_amdgcn_global_load_async_to_lds_b128)
#    define HAVE_ASYNC_LDS 1
#  endif
#endif
#ifndef HAVE_ASYNC_LDS
#  define HAVE_ASYNC_LDS 0
#endif

#define AS_GLOBAL __attribute__((address_space(1)))
#define AS_LOCAL  __attribute__((address_space(3)))

// the builtin's pointee type, per hipcc diagnostic:
//   "__attribute__((__vector_size__(4 * sizeof(int)))) int __device__ *"
typedef int gv4i __attribute__((vector_size(4 * sizeof(int))));

__device__ __forceinline__ void wait_asynccnt0() {
#if defined(__has_builtin) && __has_builtin(__builtin_amdgcn_s_wait_asynccnt)
  __builtin_amdgcn_s_wait_asynccnt(0);
#else
  asm volatile("s_wait_asynccnt 0x0" ::: "memory");
#endif
}

// ---------------- WMMA types ----------------
typedef __bf16 v16bf __attribute__((ext_vector_type(16)));
typedef float  v8f   __attribute__((ext_vector_type(8)));

union Frag16 { v16bf v; uint4 q[2]; };

__device__ __forceinline__ unsigned short f2bf(float f) {
  unsigned int u = __float_as_uint(f);
  u += 0x7FFFu + ((u >> 16) & 1u);          // round-to-nearest-even
  return (unsigned short)(u >> 16);
}

// =====================================================================
// LayerNorm: fp32 [row,1024] -> bf16 [row,1024]
// =====================================================================
__global__ __launch_bounds__(256)
void ln_kernel(const float* __restrict__ x, const float* __restrict__ g,
               const float* __restrict__ b, unsigned short* __restrict__ out)
{
  __shared__ float red[256];
  const int row = blockIdx.x;
  const int t = threadIdx.x;
  const float* xr = x + (size_t)row * D_MODEL;
  float4 v = *(const float4*)(xr + t * 4);

  float s = v.x + v.y + v.z + v.w;
  red[t] = s; __syncthreads();
  for (int st = 128; st > 0; st >>= 1) { if (t < st) red[t] += red[t + st]; __syncthreads(); }
  float mean = red[0] * (1.0f / D_MODEL);
  __syncthreads();

  float d0 = v.x - mean, d1 = v.y - mean, d2 = v.z - mean, d3 = v.w - mean;
  red[t] = d0*d0 + d1*d1 + d2*d2 + d3*d3; __syncthreads();
  for (int st = 128; st > 0; st >>= 1) { if (t < st) red[t] += red[t + st]; __syncthreads(); }
  float var = red[0] * (1.0f / D_MODEL);
  float rs = rsqrtf(var + 1e-6f);

  unsigned short* o = out + (size_t)row * D_MODEL + t * 4;
  int c0 = t * 4;
  o[0] = f2bf(d0 * rs * g[c0+0] + b[c0+0]);
  o[1] = f2bf(d1 * rs * g[c0+1] + b[c0+1]);
  o[2] = f2bf(d2 * rs * g[c0+2] + b[c0+2]);
  o[3] = f2bf(d3 * rs * g[c0+3] + b[c0+3]);
}

// =====================================================================
// elementwise fp32 -> bf16 cast
// =====================================================================
__global__ __launch_bounds__(256)
void cast_bf_kernel(const float* __restrict__ in, unsigned short* __restrict__ out, int n)
{
  int i = blockIdx.x * 256 + threadIdx.x;
  if (i < n) out[i] = f2bf(in[i]);
}

// =====================================================================
// GEMM: C[M,N] = A_bf16[M,K] * W_f32[N,K]^T + bias (+res) (relu) -> f32|bf16
// block 256 (8 waves), tile BM=128 BN=128 BK=32, wave tile 32x64 (2x4 WMMA)
// A tile staged via async global->LDS when available (pure bf16 copy);
// W tile converted fp32->bf16 through VGPRs (overlaps with async copy).
// =====================================================================
template<bool RELU, bool BF16OUT>
__global__ __launch_bounds__(256)
void gemm_kernel(const unsigned short* __restrict__ A,
                 const float* __restrict__ W,
                 const float* __restrict__ bias,
                 const float* __restrict__ res,
                 void* __restrict__ Cout,
                 int N, int Kdim)
{
  __shared__ unsigned short As[128 * 40];   // stride 40 halves = 80B (16B aligned rows)
  __shared__ unsigned short Bs[128 * 40];

  const int tid  = threadIdx.x;
  const int wv   = tid >> 5, lane = tid & 31;
  const int m16  = lane & 15, hh = lane >> 4;
  const int m0   = blockIdx.y * 128, n0 = blockIdx.x * 128;
  const int wm   = (wv & 3) * 32, wn = (wv >> 2) * 64;

  v8f C[2][4];
  #pragma unroll
  for (int i = 0; i < 2; ++i)
    #pragma unroll
    for (int j = 0; j < 4; ++j) { v8f z = {}; C[i][j] = z; }

  const int srow = tid >> 1, skg = (tid & 1) * 16;   // 128 rows x 32 halves, 2 thr/row

  for (int kk = 0; kk < Kdim; kk += 32) {
    // ---- stage A (pure bf16 copy: async DMA to LDS if available) ----
    const unsigned short* asrc = A + (size_t)(m0 + srow) * Kdim + kk + skg;
    unsigned short* adst = As + srow * 40 + skg;
#if HAVE_ASYNC_LDS
    __builtin_amdgcn_global_load_async_to_lds_b128(
        (AS_GLOBAL gv4i*)asrc, (AS_LOCAL gv4i*)adst, 0, 0);
    __builtin_amdgcn_global_load_async_to_lds_b128(
        (AS_GLOBAL gv4i*)(asrc + 8), (AS_LOCAL gv4i*)(adst + 8), 0, 0);
#else
    ((uint4*)adst)[0] = *(const uint4*)(asrc);
    ((uint4*)adst)[1] = *(const uint4*)(asrc + 8);
#endif
    // ---- stage B (fp32 -> bf16 convert) ----
    const float* bsrc = W + (size_t)(n0 + srow) * Kdim + kk + skg;
    float4 f0 = *(const float4*)(bsrc);
    float4 f1 = *(const float4*)(bsrc + 4);
    float4 f2 = *(const float4*)(bsrc + 8);
    float4 f3 = *(const float4*)(bsrc + 12);
    uint4 p0, p1;
    p0.x = (unsigned)f2bf(f0.x) | ((unsigned)f2bf(f0.y) << 16);
    p0.y = (unsigned)f2bf(f0.z) | ((unsigned)f2bf(f0.w) << 16);
    p0.z = (unsigned)f2bf(f1.x) | ((unsigned)f2bf(f1.y) << 16);
    p0.w = (unsigned)f2bf(f1.z) | ((unsigned)f2bf(f1.w) << 16);
    p1.x = (unsigned)f2bf(f2.x) | ((unsigned)f2bf(f2.y) << 16);
    p1.y = (unsigned)f2bf(f2.z) | ((unsigned)f2bf(f2.w) << 16);
    p1.z = (unsigned)f2bf(f3.x) | ((unsigned)f2bf(f3.y) << 16);
    p1.w = (unsigned)f2bf(f3.z) | ((unsigned)f2bf(f3.w) << 16);
    uint4* bdst = (uint4*)(Bs + srow * 40 + skg);
    bdst[0] = p0;
    bdst[1] = p1;
    // prefetch next K tile toward the WGP while this one computes
    if (kk + 32 < Kdim) {
      __builtin_prefetch(asrc + 32, 0, 0);
      __builtin_prefetch(bsrc + 32, 0, 0);
    }
#if HAVE_ASYNC_LDS
    wait_asynccnt0();
#endif
    __syncthreads();

    Frag16 Af[2], Bf[4];
    #pragma unroll
    for (int i = 0; i < 2; ++i) {
      const unsigned short* p = As + (wm + i * 16 + m16) * 40;
      Af[i].q[0] = *(const uint4*)(p + hh * 8);
      Af[i].q[1] = *(const uint4*)(p + 16 + hh * 8);
    }
    #pragma unroll
    for (int j = 0; j < 4; ++j) {
      const unsigned short* p = Bs + (wn + j * 16 + m16) * 40;
      Bf[j].q[0] = *(const uint4*)(p + hh * 8);
      Bf[j].q[1] = *(const uint4*)(p + 16 + hh * 8);
    }
    #pragma unroll
    for (int i = 0; i < 2; ++i)
      #pragma unroll
      for (int j = 0; j < 4; ++j)
        C[i][j] = __builtin_amdgcn_wmma_f32_16x16x32_bf16(
            false, Af[i].v, false, Bf[j].v, (short)0, C[i][j], false, false);
    __syncthreads();
  }

  // epilogue: bias (+res)(relu) -> store
  #pragma unroll
  for (int i = 0; i < 2; ++i) {
    #pragma unroll
    for (int j = 0; j < 4; ++j) {
      const int col = n0 + wn + j * 16 + m16;
      const float bi = bias[col];
      #pragma unroll
      for (int r = 0; r < 8; ++r) {
        const int rowg = m0 + wm + i * 16 + r + hh * 8;
        const size_t idx = (size_t)rowg * N + col;
        float v = C[i][j][r] + bi;
        if (res) v += res[idx];
        if (RELU) v = fmaxf(v, 0.0f);
        if (BF16OUT) ((unsigned short*)Cout)[idx] = f2bf(v);
        else         ((float*)Cout)[idx] = v;
      }
    }
  }
}

// =====================================================================
// Flash attention, one (b,h,64-q-rows) block, 4 waves x 16 q rows.
// 32 keys per iteration, online softmax, T5 rel bias via LDS table,
// optional causal. Writes bf16 attention output [B*Tq, 1024].
// =====================================================================
__global__ __launch_bounds__(128)
void attn_kernel(const float* __restrict__ Q, int ldq,
                 const float* __restrict__ K, int ldk,
                 const float* __restrict__ V, int ldv,
                 const float* __restrict__ rel,
                 unsigned short* __restrict__ O, int ldo,
                 int Tq, int Tk, int causal)
{
  __shared__ unsigned short Qs[64 * 72];     // [qrow][d]   (stride 144B)
  __shared__ unsigned short Ks[32 * 72];     // [key][d]
  __shared__ unsigned short Vt[64 * 40];     // [d][key]  (transposed)
  __shared__ unsigned short Ps[4 * 16 * 40]; // per-wave P tile [row][key32]
  __shared__ float btab[1023];               // rel bias vs (k - q + 511)

  const int b = blockIdx.z, h = blockIdx.y;
  const int qb0 = blockIdx.x * 64;
  const int tid = threadIdx.x, wv = tid >> 5, lane = tid & 31;
  const int m16 = lane & 15, hh = lane >> 4;

  // ---- precompute rel-bias table: bias depends only on rp = k - q ----
  for (int i = tid; i < 1023; i += 128) {
    const int rp = i - 511;
    const int bucket = (rp < 0) ? ((-rp < MDIST-1) ? -rp : MDIST-1)
                                : (((rp < MDIST-1) ? rp : MDIST-1) + MDIST);
    btab[i] = rel[bucket * NHEAD + h];
  }

  // ---- stage Q block (64 x 64) as bf16, pre-scaled by 1/sqrt(hd) ----
  {
    int row = tid >> 1;
    int dg = (tid & 1) * 32;
    const float* src = Q + ((size_t)(b * Tq + qb0 + row)) * ldq + h * HDIM + dg;
    unsigned short* dst = Qs + row * 72 + dg;
    #pragma unroll
    for (int j = 0; j < 8; ++j) {
      float4 f = *(const float4*)(src + j * 4);
      dst[j*4+0] = f2bf(f.x * 0.125f); dst[j*4+1] = f2bf(f.y * 0.125f);
      dst[j*4+2] = f2bf(f.z * 0.125f); dst[j*4+3] = f2bf(f.w * 0.125f);
    }
  }
  __syncthreads();

  const int qw = qb0 + wv * 16;           // this wave's first q row
  Frag16 Qa[2];
  #pragma unroll
  for (int f = 0; f < 2; ++f) {
    const unsigned short* p = Qs + (wv * 16 + m16) * 72 + f * 32;
    Qa[f].q[0] = *(const uint4*)(p + hh * 8);
    Qa[f].q[1] = *(const uint4*)(p + 16 + hh * 8);
  }

  v8f Of[4];
  #pragma unroll
  for (int d = 0; d < 4; ++d) { v8f z = {}; Of[d] = z; }
  float mrow[8], lrow[8];
  #pragma unroll
  for (int r = 0; r < 8; ++r) { mrow[r] = -1e30f; lrow[r] = 0.0f; }

  const int kend = causal ? (qb0 + 64) : Tk;

  for (int kk = 0; kk < kend; kk += 32) {
    // ---- stage K tile [32][64] and V^T tile [64][32] ----
    {
      int key = tid >> 2, dg = (tid & 3) * 16;
      const float* sk = K + ((size_t)(b * Tk + kk + key)) * ldk + h * HDIM + dg;
      unsigned short* dk = Ks + key * 72 + dg;
      #pragma unroll
      for (int j = 0; j < 4; ++j) {
        float4 f = *(const float4*)(sk + j * 4);
        dk[j*4+0] = f2bf(f.x); dk[j*4+1] = f2bf(f.y);
        dk[j*4+2] = f2bf(f.z); dk[j*4+3] = f2bf(f.w);
      }
      const float* sv = V + ((size_t)(b * Tk + kk + key)) * ldv + h * HDIM + dg;
      #pragma unroll
      for (int j = 0; j < 16; ++j)
        Vt[(dg + j) * 40 + key] = f2bf(sv[j]);
    }
    __syncthreads();

    if (!causal || kk <= qw + 15) {        // wave-uniform guard (EXEC all-ones)
      // ---- S = Q K^T (16 x 32), Q already pre-scaled ----
      v8f Sc[2];
      #pragma unroll
      for (int kt = 0; kt < 2; ++kt) {
        v8f acc = {};
        #pragma unroll
        for (int f = 0; f < 2; ++f) {
          Frag16 Kb;
          const unsigned short* p = Ks + (kt * 16 + m16) * 72 + f * 32;
          Kb.q[0] = *(const uint4*)(p + hh * 8);
          Kb.q[1] = *(const uint4*)(p + 16 + hh * 8);
          acc = __builtin_amdgcn_wmma_f32_16x16x32_bf16(
              false, Qa[f].v, false, Kb.v, (short)0, acc, false, false);
        }
        Sc[kt] = acc;
      }

      // ---- rel bias (LDS table) + causal mask, row max ----
      float rmax[8];
      #pragma unroll
      for (int r = 0; r < 8; ++r) {
        const int qg = qw + r + hh * 8;
        const int base = kk + m16 - qg + 511;
        float vmax = -1e30f;
        #pragma unroll
        for (int kt = 0; kt < 2; ++kt) {
          const int kg = kk + kt * 16 + m16;
          float s = Sc[kt][r] + btab[base + kt * 16];
          if (causal && kg > qg) s = -1e30f;
          Sc[kt][r] = s;
          vmax = fmaxf(vmax, s);
        }
        #pragma unroll
        for (int m = 1; m < 16; m <<= 1) vmax = fmaxf(vmax, __shfl_xor(vmax, m, 32));
        rmax[r] = vmax;
      }

      // ---- online softmax update, write P to LDS ----
      #pragma unroll
      for (int r = 0; r < 8; ++r) {
        const float mnew = fmaxf(mrow[r], rmax[r]);
        const float corr = __expf(mrow[r] - mnew);
        mrow[r] = mnew;
        lrow[r] *= corr;
        #pragma unroll
        for (int d = 0; d < 4; ++d) Of[d][r] *= corr;
        const float p0 = __expf(Sc[0][r] - mnew);
        const float p1 = __expf(Sc[1][r] - mnew);
        unsigned short* pr = Ps + (wv * 16 + r + hh * 8) * 40;
        pr[m16]      = f2bf(p0);
        pr[16 + m16] = f2bf(p1);
        float ps = p0 + p1;
        #pragma unroll
        for (int m = 1; m < 16; m <<= 1) ps += __shfl_xor(ps, m, 32);
        lrow[r] += ps;
      }
      asm volatile("s_wait_dscnt 0x0" ::: "memory");

      // ---- O += P V ----
      Frag16 Pa;
      {
        const unsigned short* p = Ps + (wv * 16 + m16) * 40;
        Pa.q[0] = *(const uint4*)(p + hh * 8);
        Pa.q[1] = *(const uint4*)(p + 16 + hh * 8);
      }
      #pragma unroll
      for (int d = 0; d < 4; ++d) {
        Frag16 Vb;
        const unsigned short* p = Vt + (d * 16 + m16) * 40;
        Vb.q[0] = *(const uint4*)(p + hh * 8);
        Vb.q[1] = *(const uint4*)(p + 16 + hh * 8);
        Of[d] = __builtin_amdgcn_wmma_f32_16x16x32_bf16(
            false, Pa.v, false, Vb.v, (short)0, Of[d], false, false);
      }
    }
    __syncthreads();
  }

  // ---- normalize and store bf16 ----
  #pragma unroll
  for (int d = 0; d < 4; ++d) {
    #pragma unroll
    for (int r = 0; r < 8; ++r) {
      const int mr = r + hh * 8;
      const float o = Of[d][r] / lrow[r];
      O[((size_t)(b * Tq + qw + mr)) * ldo + h * HDIM + d * 16 + m16] = f2bf(o);
    }
  }
}

// =====================================================================
// host orchestration
// =====================================================================
extern "C" void kernel_launch(void* const* d_in, const int* in_sizes, int n_in,
                              void* d_out, int out_size, void* d_ws, size_t ws_size,
                              hipStream_t stream)
{
  (void)in_sizes; (void)n_in; (void)out_size; (void)ws_size;

  const float* src          = (const float*)d_in[0];
  const float* tgt          = (const float*)d_in[1];
  const float* enc_in_w     = (const float*)d_in[2];
  const float* enc_in_b     = (const float*)d_in[3];
  const float* enc_out_w    = (const float*)d_in[4];
  const float* enc_out_b    = (const float*)d_in[5];
  const float* enc_ff1_w    = (const float*)d_in[6];
  const float* enc_ff1_b    = (const float*)d_in[7];
  const float* enc_ff2_w    = (const float*)d_in[8];
  const float* enc_ff2_b    = (const float*)d_in[9];
  const float* enc_n1_s     = (const float*)d_in[10];
  const float* enc_n1_b     = (const float*)d_in[11];
  const float* enc_n2_s     = (const float*)d_in[12];
  const float* enc_n2_b     = (const float*)d_in[13];
  const float* enc_rel      = (const float*)d_in[14];
  const float* dec_sa_in_w  = (const float*)d_in[15];
  const float* dec_sa_in_b  = (const float*)d_in[16];
  const float* dec_sa_out_w = (const float*)d_in[17];
  const float* dec_sa_out_b = (const float*)d_in[18];
  const float* dec_ca_in_w  = (const float*)d_in[19];
  const float* dec_ca_in_b  = (const float*)d_in[20];
  const float* dec_ca_out_w = (const float*)d_in[21];
  const float* dec_ca_out_b = (const float*)d_in[22];
  const float* dec_ff1_w    = (const float*)d_in[23];
  const float* dec_ff1_b    = (const float*)d_in[24];
  const float* dec_ff2_w    = (const float*)d_in[25];
  const float* dec_ff2_b    = (const float*)d_in[26];
  const float* dec_n1_s     = (const float*)d_in[27];
  const float* dec_n1_b     = (const float*)d_in[28];
  const float* dec_n2_s     = (const float*)d_in[29];
  const float* dec_n2_b     = (const float*)d_in[30];
  const float* dec_n3_s     = (const float*)d_in[31];
  const float* dec_n3_b     = (const float*)d_in[32];
  const float* dec_rel_self = (const float*)d_in[33];
  const float* dec_rel_cross= (const float*)d_in[34];

  const size_t MB = 1ull << 20;
  char* ws = (char*)d_ws;
  float*          xbuf   = (float*)(ws + 0 * MB);           // 16MB: enc residual / dec y
  float*          memf   = (float*)(ws + 16 * MB);          // 16MB: encoder memory
  unsigned short* membf  = (unsigned short*)(ws + 32 * MB); // 8MB
  unsigned short* lnbf   = (unsigned short*)(ws + 40 * MB); // 8MB
  unsigned short* attnbf = (unsigned short*)(ws + 48 * MB); // 8MB
  char*           big    = ws + 56 * MB;                    // 48MB
  float*          qkv    = (float*)big;                     // [4096,3072]
  float*          qbuf   = (float*)big;                     // [4096,1024]
  float*          kvbuf  = (float*)(big + 16 * MB);         // [4096,2048]
  unsigned short* ff1    = (unsigned short*)big;            // [4096,4096] bf16

  const dim3 B256(256), B128(128);
  const dim3 LNG(MROWS);
  const dim3 ATG(SEQ / 64, NHEAD, BSZ);
  const dim3 G3072(3072 / 128, MROWS / 128);
  const dim3 G2048(2048 / 128, MROWS / 128);
  const dim3 G1024(1024 / 128, MROWS / 128);
  const dim3 G4096(4096 / 128, MROWS / 128);

  // ================= encoder layer =================
  ln_kernel<<<LNG, B256, 0, stream>>>(src, enc_n1_s, enc_n1_b, lnbf);
  gemm_kernel<false,false><<<G3072, B256, 0, stream>>>(lnbf, enc_in_w, enc_in_b, nullptr, qkv, 3072, 1024);
  attn_kernel<<<ATG, B128, 0, stream>>>(qkv, 3072, qkv + 1024, 3072, qkv + 2048, 3072,
                                        enc_rel, attnbf, D_MODEL, SEQ, SEQ, 0);
  gemm_kernel<false,false><<<G1024, B256, 0, stream>>>(attnbf, enc_out_w, enc_out_b, src, xbuf, 1024, 1024);
  ln_kernel<<<LNG, B256, 0, stream>>>(xbuf, enc_n2_s, enc_n2_b, lnbf);
  gemm_kernel<true ,true ><<<G4096, B256, 0, stream>>>(lnbf, enc_ff1_w, enc_ff1_b, nullptr, ff1, 4096, 1024);
  gemm_kernel<false,false><<<G1024, B256, 0, stream>>>(ff1, enc_ff2_w, enc_ff2_b, xbuf, memf, 1024, 4096);
  cast_bf_kernel<<<(MROWS * D_MODEL) / 256, B256, 0, stream>>>(memf, membf, MROWS * D_MODEL);

  // ================= decoder layer =================
  ln_kernel<<<LNG, B256, 0, stream>>>(tgt, dec_n1_s, dec_n1_b, lnbf);
  gemm_kernel<false,false><<<G3072, B256, 0, stream>>>(lnbf, dec_sa_in_w, dec_sa_in_b, nullptr, qkv, 3072, 1024);
  attn_kernel<<<ATG, B128, 0, stream>>>(qkv, 3072, qkv + 1024, 3072, qkv + 2048, 3072,
                                        dec_rel_self, attnbf, D_MODEL, SEQ, SEQ, 1);
  gemm_kernel<false,false><<<G1024, B256, 0, stream>>>(attnbf, dec_sa_out_w, dec_sa_out_b, tgt, xbuf, 1024, 1024);

  ln_kernel<<<LNG, B256, 0, stream>>>(xbuf, dec_n2_s, dec_n2_b, lnbf);
  gemm_kernel<false,false><<<G1024, B256, 0, stream>>>(lnbf, dec_ca_in_w, dec_ca_in_b, nullptr, qbuf, 1024, 1024);
  gemm_kernel<false,false><<<G2048, B256, 0, stream>>>(membf, dec_ca_in_w + (size_t)1024 * 1024,
                                                       dec_ca_in_b + 1024, nullptr, kvbuf, 2048, 1024);
  attn_kernel<<<ATG, B128, 0, stream>>>(qbuf, 1024, kvbuf, 2048, kvbuf + 1024, 2048,
                                        dec_rel_cross, attnbf, D_MODEL, SEQ, SEQ, 0);
  gemm_kernel<false,false><<<G1024, B256, 0, stream>>>(attnbf, dec_ca_out_w, dec_ca_out_b, xbuf, xbuf, 1024, 1024);

  ln_kernel<<<LNG, B256, 0, stream>>>(xbuf, dec_n3_s, dec_n3_b, lnbf);
  gemm_kernel<true ,true ><<<G4096, B256, 0, stream>>>(lnbf, dec_ff1_w, dec_ff1_b, nullptr, ff1, 4096, 1024);
  gemm_kernel<false,false><<<G1024, B256, 0, stream>>>(ff1, dec_ff2_w, dec_ff2_b, xbuf, (float*)d_out, 1024, 4096);
}
